// capsule_87840671138361
// MI455X (gfx1250) — matrix operand order repaired and checked by reading, not compiled, not run on hardware
//
#include <hip/hip_runtime.h>

#define N_ 8
#define L_ 2048
#define D_ 512
#define DECAY_ 0.2f
#define QT 32
#define KT 64
#define RSQ 260   // Qs/Ks row stride in dwords (256 data + 4 pad) == TDM pad 256+4
#define VTS 36    // Vt row stride in dwords (32 data + 4 pad)
#define PBS 36    // Pb row stride in dwords (32 data + 4 pad)

typedef __attribute__((ext_vector_type(16))) __bf16 v16bf;
typedef __attribute__((ext_vector_type(8)))  float  v8f;
typedef __attribute__((ext_vector_type(4)))  unsigned int v4u;
typedef __attribute__((ext_vector_type(8)))  int v8i;
typedef __attribute__((ext_vector_type(4)))  int v4i;

union ABf { v16bf v; unsigned int u[8]; };
union CFrag { v8f v; float f[8]; };

__device__ inline unsigned short f2bf_bits(float f) {
    union { float f; unsigned int u; } x; x.f = f;
    unsigned int r = (x.u + 0x7FFFu + ((x.u >> 16) & 1u)) >> 16;
    return (unsigned short)r;
}

__device__ inline void load_frag_A(ABf& a, const unsigned int* base1) {
    uint4 g0 = *(const uint4*)(base1);
    uint4 g1 = *(const uint4*)(base1 + 4);
    a.u[0] = g0.x; a.u[1] = g0.y; a.u[2] = g0.z; a.u[3] = g0.w;
    a.u[4] = g1.x; a.u[5] = g1.y; a.u[6] = g1.z; a.u[7] = g1.w;
}
__device__ inline void load_frag_B(ABf& b, const unsigned int* base) {
    uint4 g0 = *(const uint4*)(base);
    uint4 g1 = *(const uint4*)(base + 4);
    b.u[0] = g0.x; b.u[1] = g0.y; b.u[2] = g0.z; b.u[3] = g0.w;
    b.u[4] = g1.x; b.u[5] = g1.y; b.u[6] = g1.z; b.u[7] = g1.w;
}

// TDM: stream a [rows x 512] bf16 tile (row-major, row stride 512 elements)
// into LDS at ldsByteAddr, with 16B padding after every 1024B row
// (pad_interval=7 -> 256 dwords, pad_amount=3 -> 4 dwords) => RSQ=260 rows.
__device__ inline void tdm_load_tile(unsigned int ldsByteAddr,
                                     const void* gaddr, int rows) {
    unsigned long long ga = (unsigned long long)gaddr;
    v4u g0;
    g0.s0 = 1u;                                            // count=1, user mode
    g0.s1 = ldsByteAddr;                                   // lds_addr
    g0.s2 = (unsigned int)(ga & 0xFFFFFFFFu);              // global_addr[31:0]
    g0.s3 = (unsigned int)((ga >> 32) & 0x01FFFFFFu)       // global_addr[56:32]
          | (2u << 30);                                    // type = 2 (image)
    v8i g1;
    g1.s0 = (int)((1u << 16)      // data_size = 2 bytes
                | (1u << 20)      // pad_enable
                | (7u << 22)      // pad_interval: 256 dwords
                | (3u << 25));    // pad_amount: 4 dwords
    g1.s1 = (int)(512u << 16);    // tensor_dim0 = 512 (bits 79:48, low part)
    g1.s2 = (int)(2048u << 16);   // tensor_dim1 = 2048 (bits 111:80, low part)
    g1.s3 = (int)(512u << 16);    // tile_dim0 = 512 (bits 127:112)
    g1.s4 = rows;                 // tile_dim1 (bits 143:128)
    g1.s5 = 512;                  // tensor_dim0_stride = 512 elements
    g1.s6 = 0;
    g1.s7 = 0;
    v4i z4 = {0, 0, 0, 0};
    v8i z8 = {0, 0, 0, 0, 0, 0, 0, 0};
    __builtin_amdgcn_tensor_load_to_lds(g0, g1, z4, z4, z8, 0);
}

// ---------------------------------------------------------------------------
// Kernel 1: causal depthwise convs + 3-iteration dynamic routing (M=2).
// ---------------------------------------------------------------------------
__global__ void prep_routing_kernel(const float* __restrict__ x,
                                    const float* __restrict__ w3,
                                    const float* __restrict__ w5,
                                    unsigned short* __restrict__ vS,
                                    unsigned short* __restrict__ vT) {
    __shared__ float red0[256], red1[256], red2[256];
    int pos = blockIdx.x;            // n*L + l
    int l   = pos % L_;
    int tid = threadIdx.x;
    const int CPT = 2;

    float u0s[CPT], u1s[CPT], u0t[CPT], u1t[CPT];
#pragma unroll
    for (int c = 0; c < CPT; ++c) {
        int d = tid * CPT + c;
        float xw[5];
#pragma unroll
        for (int j = 0; j < 5; ++j) {
            int li = l - 4 + j;
            xw[j] = (li >= 0) ? x[((size_t)(pos - l + li)) * D_ + d] : 0.0f;
        }
        float t5 = 0.f;
#pragma unroll
        for (int j = 0; j < 5; ++j) t5 += xw[j] * w5[d * 5 + j];
        float t3 = 0.f;
#pragma unroll
        for (int j = 0; j < 3; ++j) t3 += xw[j + 2] * w3[d * 3 + j];
        float x0 = xw[4];
        u0s[c] = x0 - t3; u1s[c] = x0 - t5;
        u0t[c] = t3;      u1t[c] = t5;
    }

    for (int which = 0; which < 2; ++which) {
        const float* u0 = (which == 0) ? u0s : u0t;
        const float* u1 = (which == 0) ? u1s : u1t;
        unsigned short* outp = (which == 0) ? vS : vT;
        float b0 = 0.f, b1 = 0.f;
        float vout[CPT];
        for (int it = 0; it < 3; ++it) {
            float c0 = 1.0f / (1.0f + __expf(b1 - b0));
            float c1 = 1.0f - c0;
            float s[CPT], pss = 0.f, pd0 = 0.f, pd1 = 0.f;
#pragma unroll
            for (int c = 0; c < CPT; ++c) {
                s[c] = c0 * u0[c] + c1 * u1[c];
                pss += s[c] * s[c];
                pd0 += u0[c] * s[c];
                pd1 += u1[c] * s[c];
            }
            __syncthreads();
            red0[tid] = pss; red1[tid] = pd0; red2[tid] = pd1;
            __syncthreads();
            for (int off = 128; off > 0; off >>= 1) {
                if (tid < off) {
                    red0[tid] += red0[tid + off];
                    red1[tid] += red1[tid + off];
                    red2[tid] += red2[tid + off];
                }
                __syncthreads();
            }
            float ss = red0[0], dd0 = red1[0], dd1 = red2[0];
            float nrm  = sqrtf(ss);
            float coef = (ss / (1.0f + ss)) / (nrm + 1e-9f);
            if (it < 2) { b0 += coef * dd0; b1 += coef * dd1; }
            else {
#pragma unroll
                for (int c = 0; c < CPT; ++c) vout[c] = coef * s[c];
            }
        }
#pragma unroll
        for (int c = 0; c < CPT; ++c)
            outp[(size_t)pos * D_ + tid * CPT + c] = f2bf_bits(vout[c]);
    }
}

// ---------------------------------------------------------------------------
// Kernel 2: fp32 -> bf16 weight conversion.
// ---------------------------------------------------------------------------
__global__ void convert_w_kernel(const float* __restrict__ w,
                                 unsigned short* __restrict__ wb, int n) {
    int i = blockIdx.x * blockDim.x + threadIdx.x;
    if (i < n) wb[i] = f2bf_bits(w[i]);
}

// ---------------------------------------------------------------------------
// Kernel 3: flash attention with causal mask + linear decay bias.
// Block = 8 waves (1 WG/WGP), query tile 32, key tile 64.
// Q/K tiles streamed by the Tensor Data Mover (hardware row padding);
// transposed V copy built from global in parallel with the TDM stream.
// ---------------------------------------------------------------------------
__global__ void __launch_bounds__(256, 1)
flash_attn_kernel(const unsigned short* __restrict__ X, // [N,L,D] bf16
                  unsigned short* __restrict__ cat,     // [N,L,2D] bf16
                  int colOff) {
    extern __shared__ char smem[];
    unsigned int* Qd = (unsigned int*)smem;                 // QT*RSQ dwords
    unsigned int* Kd = Qd + QT * RSQ;                       // KT*RSQ dwords
    unsigned int* Vt = Kd + KT * RSQ;                       // D*VTS dwords
    float* Sx   = (float*)(Vt + D_ * VTS);                  // QT*KT
    unsigned int* Pb = (unsigned int*)(Sx + QT * KT);       // QT*PBS dwords
    float* rowm = (float*)(Pb + QT * PBS);
    float* rowl = rowm + QT;
    float* rowscale = rowl + QT;

    int nb    = blockIdx.x / (L_ / QT);
    int qtile = blockIdx.x % (L_ / QT);
    int qbase = qtile * QT;
    int tid  = threadIdx.x;
    int wave = tid >> 5;
    int lane = tid & 31;
    int lr   = lane & 15;
    int hi   = lane >> 4;

    const unsigned short* Xrow = X + (size_t)nb * L_ * D_;
    const unsigned int* Xg = (const unsigned int*)Xrow;
    const int DW = D_ / 2;   // 256 dwords per row

    // Q tile via TDM (wave 0 issues and waits; barrier publishes)
    if (wave == 0) {
        tdm_load_tile(0u, Xrow + (size_t)qbase * D_, QT);
        __builtin_amdgcn_s_wait_tensorcnt(0);
    }
    if (tid < QT) { rowm[tid] = -3.0e38f; rowl[tid] = 0.f; }
    __syncthreads();

    int mts = wave >> 2;
    int nts = wave & 3;

    // hoist Q A-fragments for this wave's S row-tile (invariant over key loop)
    ABf qa[D_ / 32];
    {
        const unsigned int* arow = Qd + (16 * mts + lr) * RSQ + 8 * hi;
#pragma unroll
        for (int kc = 0; kc < D_ / 32; ++kc) load_frag_A(qa[kc], arow + kc * 16);
    }

    CFrag acc[2][4];
#pragma unroll
    for (int mt = 0; mt < 2; ++mt)
#pragma unroll
        for (int nt = 0; nt < 4; ++nt)
#pragma unroll
            for (int v = 0; v < 8; ++v) acc[mt][nt].f[v] = 0.f;

    int nkb = (qbase + QT - 1) / KT + 1;
    for (int kb = 0; kb < nkb; ++kb) {
        int kbase = kb * KT;
        __syncthreads();
        // K tile via TDM; transposed V copy built from global in parallel
        if (wave == 0)
            tdm_load_tile((unsigned int)(QT * RSQ * 4),
                          Xrow + (size_t)kbase * D_, KT);
        {
            const unsigned int* ksrc = Xg + (size_t)kbase * DW;
            for (int u = tid; u < (KT / 2) * DW; u += 256) {
                int kp = u >> 8, dp = u & 255;
                unsigned int g0 = ksrc[(2 * kp) * DW + dp];
                unsigned int g1 = ksrc[(2 * kp + 1) * DW + dp];
                unsigned int p0 = (g0 & 0xFFFFu) | (g1 << 16);
                unsigned int p1 = (g0 >> 16) | (g1 & 0xFFFF0000u);
                Vt[(2 * dp) * VTS + kp] = p0;
                Vt[(2 * dp + 1) * VTS + kp] = p1;
            }
        }
        if (wave == 0) __builtin_amdgcn_s_wait_tensorcnt(0);
        __syncthreads();

        // ---- S = Q · K^T (full D per wave, one 16x16 subtile) ----
        CFrag sc;
#pragma unroll
        for (int v = 0; v < 8; ++v) sc.f[v] = 0.f;
        const unsigned int* brow = Kd + (16 * nts + lr) * RSQ + 8 * hi;
#pragma unroll
        for (int kc = 0; kc < D_ / 32; ++kc) {
            ABf b;
            load_frag_B(b, brow + kc * 16);
            sc.v = __builtin_amdgcn_wmma_f32_16x16x32_bf16(
                false, qa[kc].v, false, b.v, (short)0, sc.v, false, false);
        }
        // bias + causal mask -> LDS S (f32)
#pragma unroll
        for (int v = 0; v < 8; ++v) {
            int m = v + 8 * hi;
            int i = qbase + 16 * mts + m;
            int j = kbase + 16 * nts + lr;
            float val = (j <= i) ? (sc.f[v] - DECAY_ * (float)(i - j)) : -3.0e38f;
            Sx[(16 * mts + m) * KT + 16 * nts + lr] = val;
        }
        __syncthreads();

        // ---- online softmax (one thread per query row); pack P to bf16 ----
        if (tid < QT) {
            int r = tid;
            float mold = rowm[r], mnew = mold;
            for (int c = 0; c < KT; ++c) mnew = fmaxf(mnew, Sx[r * KT + c]);
            float scl = __expf(mold - mnew);
            float lnew = scl * rowl[r];
            for (int c = 0; c < KT; c += 2) {
                float p0 = __expf(Sx[r * KT + c] - mnew);
                float p1 = __expf(Sx[r * KT + c + 1] - mnew);
                lnew += p0 + p1;
                Pb[r * PBS + (c >> 1)] =
                    (unsigned int)f2bf_bits(p0) | ((unsigned int)f2bf_bits(p1) << 16);
            }
            rowm[r] = mnew; rowl[r] = lnew; rowscale[r] = scl;
        }
        __syncthreads();

        // ---- O = diag(scale)·O + P · V ----
#pragma unroll
        for (int mt = 0; mt < 2; ++mt) {
#pragma unroll
            for (int nt = 0; nt < 4; ++nt)
#pragma unroll
                for (int v = 0; v < 8; ++v)
                    acc[mt][nt].f[v] *= rowscale[16 * mt + 8 * hi + v];
#pragma unroll
            for (int kk = 0; kk < KT / 32; ++kk) {
                ABf a;
                load_frag_A(a, Pb + (16 * mt + lr) * PBS + kk * 16 + 8 * hi);
#pragma unroll
                for (int nt = 0; nt < 4; ++nt) {
                    ABf b;
                    int col = 64 * wave + 16 * nt + lr;
                    load_frag_B(b, Vt + col * VTS + kk * 16 + 8 * hi);
                    acc[mt][nt].v = __builtin_amdgcn_wmma_f32_16x16x32_bf16(
                        false, a.v, false, b.v, (short)0, acc[mt][nt].v, false, false);
                }
            }
        }
    }
    __syncthreads();

    // ---- normalize and write bf16 into cat buffer ----
#pragma unroll
    for (int mt = 0; mt < 2; ++mt)
#pragma unroll
        for (int nt = 0; nt < 4; ++nt)
#pragma unroll
            for (int v = 0; v < 8; ++v) {
                int r = 16 * mt + 8 * hi + v;
                float o = acc[mt][nt].f[v] / rowl[r];
                int row = qbase + r;
                int col = colOff + 64 * wave + 16 * nt + lr;
                cat[((size_t)nb * L_ + row) * (2 * D_) + col] = f2bf_bits(o);
            }
}

// ---------------------------------------------------------------------------
// Kernel 4: fusion GEMM  out[m,n] = sum_k cat[m,k]*W[n,k] + b[n]  (fp32 out)
// One wave per 16x64 output strip: A fragment reused by 4 WMMAs.
// ---------------------------------------------------------------------------
__global__ void fusion_gemm_kernel(const unsigned short* __restrict__ cat,
                                   const unsigned short* __restrict__ Wb,
                                   const float* __restrict__ bias,
                                   float* __restrict__ out) {
    int gw   = (blockIdx.x * blockDim.x + threadIdx.x) >> 5;
    int lane = threadIdx.x & 31;
    int lr = lane & 15, hi = lane >> 4;
    int ng    = gw & 7;
    int mtile = gw >> 3;
    const int KW = D_;      // K in dwords

    CFrag c[4];
#pragma unroll
    for (int nt = 0; nt < 4; ++nt)
#pragma unroll
        for (int v = 0; v < 8; ++v) c[nt].f[v] = 0.f;

    const unsigned int* arow = (const unsigned int*)cat + (size_t)(16 * mtile + lr) * KW + 8 * hi;
    const unsigned int* brow0 = (const unsigned int*)Wb + (size_t)(16 * 4 * ng + lr) * KW + 8 * hi;
    for (int kc = 0; kc < 2 * D_ / 32; ++kc) {
        ABf a;
        load_frag_A(a, arow + kc * 16);
#pragma unroll
        for (int nt = 0; nt < 4; ++nt) {
            ABf b;
            load_frag_B(b, brow0 + (size_t)(16 * nt) * KW + kc * 16);
            c[nt].v = __builtin_amdgcn_wmma_f32_16x16x32_bf16(
                false, a.v, false, b.v, (short)0, c[nt].v, false, false);
        }
    }
#pragma unroll
    for (int nt = 0; nt < 4; ++nt)
#pragma unroll
        for (int v = 0; v < 8; ++v) {
            int m = 16 * mtile + 8 * hi + v;
            int n = 16 * (4 * ng + nt) + lr;
            out[(size_t)m * D_ + n] = c[nt].f[v] + bias[n];
        }
}

// ---------------------------------------------------------------------------
extern "C" void kernel_launch(void* const* d_in, const int* in_sizes, int n_in,
                              void* d_out, int out_size, void* d_ws, size_t ws_size,
                              hipStream_t stream) {
    const float* x  = (const float*)d_in[0];
    const float* w3 = (const float*)d_in[1];
    const float* w5 = (const float*)d_in[2];
    const float* fw = (const float*)d_in[3];
    const float* fb = (const float*)d_in[4];
    float* out = (float*)d_out;

    size_t SZ = (size_t)N_ * L_ * D_;
    unsigned short* vS  = (unsigned short*)d_ws;      // [N,L,D] bf16
    unsigned short* vT  = vS + SZ;                    // [N,L,D] bf16
    unsigned short* cat = vT + SZ;                    // [N,L,2D] bf16
    unsigned short* Wb  = cat + 2 * SZ;               // [512,1024] bf16

    prep_routing_kernel<<<N_ * L_, 256, 0, stream>>>(x, w3, w5, vS, vT);
    convert_w_kernel<<<(D_ * 2 * D_ + 255) / 256, 256, 0, stream>>>(fw, Wb, D_ * 2 * D_);

    size_t smem = (size_t)QT * RSQ * 4 + (size_t)KT * RSQ * 4      // Qd, Kd
                + (size_t)D_ * VTS * 4                              // Vt
                + (size_t)QT * KT * 4                               // Sx
                + (size_t)QT * PBS * 4                              // Pb
                + 3 * QT * 4;                                       // row stats
    (void)hipFuncSetAttribute((const void*)flash_attn_kernel,
                              hipFuncAttributeMaxDynamicSharedMemorySize, (int)smem);

    flash_attn_kernel<<<N_ * (L_ / QT), 256, smem, stream>>>(vS, cat, 0);
    flash_attn_kernel<<<N_ * (L_ / QT), 256, smem, stream>>>(vT, cat, D_);

    int fusion_blocks = ((N_ * L_ / 16) * 8) / 8;  // 1024 blocks
    fusion_gemm_kernel<<<fusion_blocks, 256, 0, stream>>>(cat, Wb, fb, out);
}
